// Map_79748952752358
// MI455X (gfx1250) — compile-verified, matrix operand order checked by reading, not run on hardware
//
#include <hip/hip_runtime.h>
#include <cstddef>

// ---------------------------------------------------------------------------
// Problem dims (from reference): B=16, S=40, V=1004, E=256, HID=256,
// C=1024, R=256, HW=32.  RC = R*C = 262144.  P = HW*HW = 1024.
// ---------------------------------------------------------------------------

typedef _Float16 v16h __attribute__((ext_vector_type(16)));
typedef _Float16 v8h  __attribute__((ext_vector_type(8)));
typedef float    v8f  __attribute__((ext_vector_type(8)));
typedef float    v4f  __attribute__((ext_vector_type(4)));

__device__ __forceinline__ v8f wmma16(v16h a, v16h b, v8f c) {
  // D = A(16x32 f16) x B(32x16 f16) + C(16x16 f32)
  return __builtin_amdgcn_wmma_f32_16x16x32_f16(
      /*neg_a=*/false, a, /*neg_b=*/false, b,
      /*c_mod=*/(short)0, c, /*reuse_a=*/false, /*reuse_b=*/false);
}

// A/B fragment loader from a contiguous f16 row (length >= k0+32).
// Lane layout (ISA 7.12.2, 16-bit 16x32): lane holds K pairs; half = lane>>4
// selects K offset +8 within each 16-K group.
__device__ __forceinline__ v16h load_f16_row(const _Float16* p, int k0, int half) {
  v8h lo = *(const v8h*)(p + k0 + half * 8);
  v8h hi = *(const v8h*)(p + k0 + 16 + half * 8);
  v16h r;
#pragma unroll
  for (int i = 0; i < 8; ++i) { r[i] = lo[i]; r[i + 8] = hi[i]; }
  return r;
}

// Same fragment but source row is fp32: load vectorized, cvt in-register.
__device__ __forceinline__ v16h load_f32_row_cvt(const float* p, int k0, int half) {
  v4f a0 = *(const v4f*)(p + k0 + half * 8);
  v4f a1 = *(const v4f*)(p + k0 + half * 8 + 4);
  v4f a2 = *(const v4f*)(p + k0 + 16 + half * 8);
  v4f a3 = *(const v4f*)(p + k0 + 16 + half * 8 + 4);
  v16h r;
#pragma unroll
  for (int i = 0; i < 4; ++i) {
    r[i]      = (_Float16)a0[i];
    r[i + 4]  = (_Float16)a1[i];
    r[i + 8]  = (_Float16)a2[i];
    r[i + 12] = (_Float16)a3[i];
  }
  return r;
}

__device__ __forceinline__ float sigm(float x) { return 1.0f / (1.0f + __expf(-x)); }

// ---------------------------------------------------------------------------
// Kernel 0: convert W_hh (1024x256 fp32) -> f16 once per call.
// ---------------------------------------------------------------------------
__global__ __launch_bounds__(256) void cvt_whh_kernel(const float* __restrict__ w,
                                                      _Float16* __restrict__ o) {
  int i = (blockIdx.x * 256 + threadIdx.x) * 4;
#pragma unroll
  for (int j = 0; j < 4; ++j) o[i + j] = (_Float16)w[i + j];
}

// ---------------------------------------------------------------------------
// Kernel 1: Xg[t] = x_t @ W_ih^T  for all 40 steps (640x1024 fp32 out).
// One block per timestep t; 8 waves x 8 column-tiles each.
// ---------------------------------------------------------------------------
__global__ __launch_bounds__(256) void xpart_kernel(const int* __restrict__ idx,
                                                    const float* __restrict__ emb,
                                                    const float* __restrict__ w_ih,
                                                    float* __restrict__ xg) {
  const int tid = threadIdx.x, wave = tid >> 5, lane = tid & 31;
  const int n = lane & 15, half = lane >> 4;
  const int t = blockIdx.x;  // 0..39

  // A = x_t (16 batch x 256), gathered via embedding lookup, cvt fp32->f16.
  const int b = n;  // A row = batch
  const int e = idx[b * 40 + t];
  const float* xrow = emb + (size_t)e * 256;
  v16h A[8];
#pragma unroll
  for (int kb = 0; kb < 8; ++kb) A[kb] = load_f32_row_cvt(xrow, kb * 32, half);

#pragma unroll 1
  for (int c = 0; c < 8; ++c) {
    const int col = (wave * 8 + c) * 16 + n;  // gate column 0..1023
    const float* wrow = w_ih + (size_t)col * 256;
    v8f acc = {};
#pragma unroll
    for (int kb = 0; kb < 8; ++kb)
      acc = wmma16(A[kb], load_f32_row_cvt(wrow, kb * 32, half), acc);
#pragma unroll
    for (int s = 0; s < 8; ++s) {
      const int batch = s + 8 * half;
      xg[(size_t)(t * 16 + batch) * 1024 + col] = acc[s];
    }
  }
}

// ---------------------------------------------------------------------------
// Kernel 2: sequential LSTM recurrence + e2d projection.
// One workgroup, 32 waves.  Wave w: ub = w>>1 (u-block of 16 hidden units),
// pair = w&1 (0 -> gates i,f ; 1 -> gates g,o).  h lives in LDS (f16),
// c lives in registers (D-tile layout) on pair==0 waves.
// ---------------------------------------------------------------------------
__global__ __launch_bounds__(1024) void lstm_rec_kernel(
    const int* __restrict__ len, const float* __restrict__ xg,
    const _Float16* __restrict__ whh, const float* __restrict__ b_ih,
    const float* __restrict__ b_hh, const float* __restrict__ e2d_w,
    const float* __restrict__ e2d_b, _Float16* __restrict__ instr_out) {
  __shared__ _Float16 hs[16 * 256];       // h state, [batch][u], f16  (8 KB)
  __shared__ float gsh[2][16][256];       // g,o tiles per ub, [q][ub][b*16+u] (32 KB)

  const int tid = threadIdx.x, wave = tid >> 5, lane = tid & 31;
  const int n = lane & 15, half = lane >> 4;
  const int ub = wave >> 1, pair = wave & 1;
  const int ga = pair * 2, gb = ga + 1;
  const int col0 = ga * 256 + ub * 16 + n;
  const int col1 = gb * 256 + ub * 16 + n;
  const float bias0 = b_ih[col0] + b_hh[col0];
  const float bias1 = b_ih[col1] + b_hh[col1];
  const _Float16* wrow0 = whh + (size_t)col0 * 256;
  const _Float16* wrow1 = whh + (size_t)col1 * 256;

  int lenr[8];
#pragma unroll
  for (int s = 0; s < 8; ++s) lenr[s] = len[s + 8 * half];

  for (int i = tid; i < 16 * 256; i += 1024) hs[i] = (_Float16)0.0f;
  v8f hreg = {}, creg = {};
  __syncthreads();

  const _Float16* hrow = &hs[(lane & 15) * 256];  // A row = batch

#pragma unroll 1
  for (int t = 0; t < 40; ++t) {
    v8f acc0 = {}, acc1 = {};
#pragma unroll
    for (int kb = 0; kb < 8; ++kb) {
      v16h a = load_f16_row(hrow, kb * 32, half);
      acc0 = wmma16(a, load_f16_row(wrow0, kb * 32, half), acc0);
      acc1 = wmma16(a, load_f16_row(wrow1, kb * 32, half), acc1);
    }
#pragma unroll
    for (int s = 0; s < 8; ++s) {
      const int batch = s + 8 * half;
      const float x0 = xg[(size_t)(t * 16 + batch) * 1024 + col0];
      const float x1 = xg[(size_t)(t * 16 + batch) * 1024 + col1];
      acc0[s] += x0 + bias0;
      acc1[s] += x1 + bias1;
    }
    if (pair == 1) {
#pragma unroll
      for (int s = 0; s < 8; ++s) {
        const int batch = s + 8 * half;
        gsh[0][ub][batch * 16 + n] = acc0[s];  // gate g
        gsh[1][ub][batch * 16 + n] = acc1[s];  // gate o
      }
    }
    __syncthreads();
    if (pair == 0) {
#pragma unroll
      for (int s = 0; s < 8; ++s) {
        const int batch = s + 8 * half;
        const float ig = acc0[s], fg = acc1[s];
        const float gg = gsh[0][ub][batch * 16 + n];
        const float og = gsh[1][ub][batch * 16 + n];
        const float cn = sigm(fg) * creg[s] + sigm(ig) * tanhf(gg);
        const float hn = sigm(og) * tanhf(cn);
        const bool sel = t < lenr[s];
        creg[s] = sel ? cn : creg[s];
        hreg[s] = sel ? hn : hreg[s];
        hs[batch * 256 + ub * 16 + n] = (_Float16)hreg[s];
      }
    }
    __syncthreads();  // h writes visible before next step's A loads
  }

  // e2d: instr = tanh(h @ e2d_w^T + e2d_b); waves 0..15 each do one 16-col tile.
  if (wave < 16) {
    const int n0 = wave * 16;
    const float* wrow = e2d_w + (size_t)(n0 + n) * 256;
    v8f acc = {};
#pragma unroll
    for (int kb = 0; kb < 8; ++kb)
      acc = wmma16(load_f16_row(hrow, kb * 32, half),
                   load_f32_row_cvt(wrow, kb * 32, half), acc);
    const float bb = e2d_b[n0 + n];
#pragma unroll
    for (int s = 0; s < 8; ++s) {
      const int batch = s + 8 * half;
      instr_out[batch * 256 + n0 + n] = (_Float16)tanhf(acc[s] + bb);
    }
  }
}

// ---------------------------------------------------------------------------
// Kernel 3: f1 = leaky_relu(instr @ lin1_w^T + lin1_b) -> f16, [b][rc].
// 256 blocks x 8 waves; wave holds A (16x256 f16) in regs, does 8 tiles.
// Memory-bound on lin1_w (268 MB fp32) -- the pipeline's HBM floor.
// ---------------------------------------------------------------------------
__global__ __launch_bounds__(256) void lin1_kernel(const _Float16* __restrict__ instr,
                                                   const float* __restrict__ w,
                                                   const float* __restrict__ bias,
                                                   _Float16* __restrict__ f1) {
  const int tid = threadIdx.x, wave = tid >> 5, lane = tid & 31;
  const int n = lane & 15, half = lane >> 4;
  const int waveId = blockIdx.x * 8 + wave;  // 0..2047

  const _Float16* arow = instr + n * 256;  // A row = batch
  v16h A[8];
#pragma unroll
  for (int kb = 0; kb < 8; ++kb) A[kb] = load_f16_row(arow, kb * 32, half);

#pragma unroll 1
  for (int tt = 0; tt < 8; ++tt) {
    const int rc0 = (waveId * 8 + tt) * 16;
    const int col = rc0 + n;
    const float* wrow = w + (size_t)col * 256;
    v8f acc = {};
#pragma unroll
    for (int kb = 0; kb < 8; ++kb)
      acc = wmma16(A[kb], load_f32_row_cvt(wrow, kb * 32, half), acc);
    const float bb = bias[col];
#pragma unroll
    for (int s = 0; s < 8; ++s) {
      float v = acc[s] + bb;
      v = v > 0.0f ? v : 0.01f * v;  // leaky_relu
      const int batch = s + 8 * half;
      f1[(size_t)batch * 262144 + col] = (_Float16)v;
    }
  }
}

// ---------------------------------------------------------------------------
// Kernel 4: fused einsum('bchw,brc->brhw') + relu + BN affine + max_r + clip.
// 1024 blocks (= 16 batches x 64 p-tiles) x 4 waves; each wave covers 4
// r-tiles over the full K=1024, then LDS max-reduction over R.
// ---------------------------------------------------------------------------
__global__ __launch_bounds__(128) void einsum_kernel(
    const float* __restrict__ feature, const _Float16* __restrict__ f1,
    const float* __restrict__ gamma, const float* __restrict__ beta,
    const float* __restrict__ mean, const float* __restrict__ var,
    float* __restrict__ out) {
  __shared__ float s_scale[256], s_shift[256];
  __shared__ float part[128];

  const int tid = threadIdx.x, wave = tid >> 5, lane = tid & 31;
  const int n = lane & 15, half = lane >> 4;
  const int b = blockIdx.x >> 6, pt = blockIdx.x & 63;
  const int p0 = pt * 16;

  for (int r = tid; r < 256; r += 128) {
    const float sc = gamma[r] * rsqrtf(var[r] + 1e-5f);
    s_scale[r] = sc;
    s_shift[r] = beta[r] - mean[r] * sc;
  }
  __syncthreads();

  const float* fb = feature + (size_t)b * 1024 * 1024;  // [c][p]
  const _Float16* f1b = f1 + (size_t)b * 262144;        // [r][c]
  const int rt0 = wave * 4;
  const int p = p0 + n;

  v8f acc[4] = {};
#pragma unroll 1
  for (int kb = 0; kb < 32; ++kb) {
    const int k0 = kb * 32;
    // B fragment from feature (K = c strided by 1024 floats; lanes coalesce over p).
    v16h bm;
#pragma unroll
    for (int j = 0; j < 8; ++j) {
      const int c = k0 + ((j >> 2) << 4) + half * 8 + ((j & 3) << 1);
      bm[2 * j]     = (_Float16)fb[(size_t)c * 1024 + p];
      bm[2 * j + 1] = (_Float16)fb[(size_t)(c + 1) * 1024 + p];
    }
#pragma unroll
    for (int rr = 0; rr < 4; ++rr) {
      const int r = (rt0 + rr) * 16 + n;  // A row = r
      acc[rr] = wmma16(load_f16_row(f1b + (size_t)r * 1024, k0, half), bm, acc[rr]);
    }
  }

  // relu -> BN affine -> per-lane max over this wave's 32 R values (same p).
  float mymax = -1e30f;
#pragma unroll
  for (int rr = 0; rr < 4; ++rr) {
#pragma unroll
    for (int s = 0; s < 8; ++s) {
      const int r = (rt0 + rr) * 16 + s + 8 * half;
      float v = acc[rr][s];
      v = v > 0.0f ? v : 0.0f;
      v = v * s_scale[r] + s_shift[r];
      mymax = fmaxf(mymax, v);
    }
  }
  part[wave * 32 + lane] = mymax;
  __syncthreads();

  if (tid < 16) {
    float m = -1e30f;
#pragma unroll
    for (int w = 0; w < 4; ++w)
      m = fmaxf(m, fmaxf(part[w * 32 + tid], part[w * 32 + tid + 16]));
    m = fminf(fmaxf(m, 0.0f), 1.0f);  // clip
    out[(size_t)b * 1024 + p0 + tid] = m;
  }
}

// ---------------------------------------------------------------------------
// Host-side launcher
// ---------------------------------------------------------------------------
extern "C" void kernel_launch(void* const* d_in, const int* in_sizes, int n_in,
                              void* d_out, int out_size, void* d_ws, size_t ws_size,
                              hipStream_t stream) {
  const float* feature = (const float*)d_in[0];
  const int*   idx     = (const int*)d_in[2];
  const int*   len     = (const int*)d_in[3];
  const float* emb     = (const float*)d_in[5];
  const float* w_ih    = (const float*)d_in[6];
  const float* w_hh    = (const float*)d_in[7];
  const float* b_ih    = (const float*)d_in[8];
  const float* b_hh    = (const float*)d_in[9];
  const float* e2d_w   = (const float*)d_in[10];
  const float* e2d_b   = (const float*)d_in[11];
  const float* lin1_w  = (const float*)d_in[12];
  const float* lin1_b  = (const float*)d_in[13];
  const float* gamma   = (const float*)d_in[14];
  const float* beta    = (const float*)d_in[15];
  const float* mean    = (const float*)d_in[16];
  const float* var     = (const float*)d_in[17];
  float* out = (float*)d_out;

  // Workspace layout (all 16B-aligned offsets):
  //   [0,        8 KB)   instr  f16 (16x256)
  //   [8 KB,    +8 MB)   f1     f16 (16x262144)
  //   [+,     +512 KB)   w_hh   f16 (1024x256)
  //   [+,    +2.56 MB)   Xg     f32 (640x1024)
  char* ws = (char*)d_ws;
  _Float16* instr16 = (_Float16*)(ws);
  _Float16* f116    = (_Float16*)(ws + 8192);
  _Float16* whh16   = (_Float16*)(ws + 8192 + 8388608);
  float*    xg      = (float*)   (ws + 8192 + 8388608 + 524288);

  cvt_whh_kernel<<<256, 256, 0, stream>>>(w_hh, whh16);
  xpart_kernel<<<40, 256, 0, stream>>>(idx, emb, w_ih, xg);
  lstm_rec_kernel<<<1, 1024, 0, stream>>>(len, xg, whh16, b_ih, b_hh,
                                          e2d_w, e2d_b, instr16);
  lin1_kernel<<<256, 256, 0, stream>>>(instr16, lin1_w, lin1_b, f116);
  einsum_kernel<<<1024, 128, 0, stream>>>(feature, f116, gamma, beta, mean, var, out);
}
